// AutoregressiveDecoder_8924942042048
// MI455X (gfx1250) — compile-verified, matrix operand order checked by reading, not run on hardware
//
#include <hip/hip_runtime.h>
#include <hip/hip_bf16.h>
#include <math.h>

// ---------------------------------------------------------------------------
// CDNA5 (gfx1250) WMMA types
// ---------------------------------------------------------------------------
typedef __attribute__((ext_vector_type(16))) __bf16 v16bf;
typedef __attribute__((ext_vector_type(8)))  float  v8f;

// float -> bf16 (round to nearest even)
static __device__ __forceinline__ unsigned short f2bf(float f) {
  unsigned int u = __float_as_uint(f);
  unsigned int r = u + 0x7FFFu + ((u >> 16) & 1u);
  return (unsigned short)(r >> 16);
}

// ---------------------------------------------------------------------------
// Async memory->LDS copy, 16 bytes per active lane (CDNA5 ISA).
// dsaddr = wave-relative LDS byte offset = low 32 bits of a flat pointer into
// __shared__. Tracked by ASYNCcnt (in-order).
// ---------------------------------------------------------------------------
static __device__ __forceinline__ void async_copy16(void* lds, const void* gptr) {
  unsigned ldsOff = (unsigned)(size_t)lds;
  asm volatile("global_load_async_to_lds_b128 %0, %1, off"
               :: "v"(ldsOff), "v"(gptr) : "memory");
}

// ---------------------------------------------------------------------------
// Packed tile layouts (per CDNA5 ISA §7.12.2, wave32):
// A (16x32 bf16, MxK):  lane = (m%16) + 16*hi,  hi = (kk>>3)&1,
//                        slot = (kk&7) + (kk>=16 ? 8 : 0),  kk = k%32
// Each lane holds 16 bf16 = 32 contiguous bytes in the packed tile.
// B (32x16 bf16, KxN):  symmetric with N on lanes 0..15.
// C/D (16x16 f32):      lane = n%16 + 16*hi ; vgpr r -> m = r + 8*hi
// ---------------------------------------------------------------------------
static __device__ __forceinline__ size_t apack_off(int m, int kglob, int Ktiles) {
  int mt = m >> 4, kt = kglob >> 5, kk = kglob & 31;
  int hi = (kk >> 3) & 1;
  int s  = (kk & 7) + ((kk & 16) ? 8 : 0);
  int lane = (m & 15) + (hi << 4);
  return ((size_t)mt * Ktiles + kt) * 512 + (size_t)lane * 16 + s;
}
static __device__ __forceinline__ size_t bpack_off(int kglob, int n, int NtPad) {
  int kt = kglob >> 5, kk = kglob & 31;
  int hi = (kk >> 3) & 1;
  int s  = (kk & 7) + ((kk & 16) ? 8 : 0);
  int nt = n >> 4;
  int lane = (n & 15) + (hi << 4);
  return ((size_t)kt * NtPad + nt) * 512 + (size_t)lane * 16 + s;
}

// ---------------------------------------------------------------------------
// Weight pack: fp32 (K x N row-major, stride ldN) -> bf16 B-tiles.
// ---------------------------------------------------------------------------
__global__ void pack_weight(const float* __restrict__ src, int rowOff, int Kcount,
                            int Ncount, int ldN, unsigned short* __restrict__ dst,
                            int NtPad, int dstKtOff, int Kpad) {
  int Npad = NtPad * 16;
  size_t total = (size_t)Kpad * Npad;
  size_t idx = (size_t)blockIdx.x * blockDim.x + threadIdx.x;
  if (idx >= total) return;
  int k = (int)(idx / Npad), n = (int)(idx % Npad);
  float v = (k < Kcount && n < Ncount) ? src[(size_t)(rowOff + k) * ldN + n] : 0.f;
  dst[bpack_off(dstKtOff * 32 + k, n, NtPad)] = f2bf(v);
}

// ---------------------------------------------------------------------------
// Activation pack: fp32 (rows x K, stride srcLd) -> bf16 A-tiles.
// ---------------------------------------------------------------------------
__global__ void pack_act(const float* __restrict__ src, int srcLd, int Kcount, int Kpad,
                         unsigned short* __restrict__ dst, int Ktiles, int dstKtOff,
                         int Mcount, int mOff, int rowMod) {
  size_t idx = (size_t)blockIdx.x * blockDim.x + threadIdx.x;
  size_t total = (size_t)Mcount * Kpad;
  if (idx >= total) return;
  int m = (int)(idx / Kpad), k = (int)(idx % Kpad);
  int dm = mOff + m;
  int row = rowMod ? (dm % rowMod) : m;
  float v = (k < Kcount) ? src[(size_t)row * srcLd + k] : 0.f;
  dst[apack_off(dm, dstKtOff * 32 + k, Ktiles)] = f2bf(v);
}

// ---------------------------------------------------------------------------
// WMMA bf16 GEMM: C[M x N] = Apacked @ Bpacked (+bias) (+addend) (relu?)
// Double-buffered LDS staging via GLOBAL_LOAD_ASYNC_TO_LDS_B128: the async
// loads for K-step kt+1 fly while kt's wmma batch executes; ASYNCcnt is
// in-order per wave, so waiting "<= newer-batch count" proves the older
// batch has landed.
// ---------------------------------------------------------------------------
template<int WM, int WN, int NT>
__global__ __launch_bounds__(WM * WN * 32)
void gemm_bf16(const unsigned short* __restrict__ Ap,
               const unsigned short* __restrict__ Bp,
               int Ktiles, int NtPadB,
               float* __restrict__ C, int ldC, int Nreal,
               const float* __restrict__ bias,
               const float* __restrict__ addend, int doRelu) {
  constexpr int NTHREADS = WM * WN * 32;
  constexpr int ACH = WM * 64;            // A 16B-chunks per K-step
  constexpr int BCH = WN * NT * 64;       // B 16B-chunks per K-step
  constexpr int STAGE = (WM + WN * NT) * 512;
  // per-wave minimum async ops issued per K-step (chunk counts are multiples
  // of 64 => floor(total/NTHREADS) lower-bounds every wave's issue count)
  constexpr int MIN_INFLIGHT = ACH / NTHREADS + BCH / NTHREADS;

  __shared__ __align__(32) unsigned short sm[2][STAGE];

  const int tid  = threadIdx.x;
  const int w    = tid >> 5;
  const int lane = tid & 31;
  const int wm   = w % WM;
  const int wn   = w / WM;
  const int mt0    = blockIdx.y * WM;
  const int ntBase = blockIdx.x * (WN * NT);

  auto stage = [&](int kt, int buf) {
    unsigned short* sA = sm[buf];
    unsigned short* sB = sm[buf] + WM * 512;
    for (int i = tid; i < ACH; i += NTHREADS) {
      int tile = i >> 6, off = i & 63;
      const uint4* g = (const uint4*)(Ap + ((size_t)(mt0 + tile) * Ktiles + kt) * 512) + off;
      async_copy16((uint4*)sA + i, g);
    }
    for (int i = tid; i < BCH; i += NTHREADS) {
      int tile = i >> 6, off = i & 63;
      const uint4* g = (const uint4*)(Bp + ((size_t)kt * NtPadB + (ntBase + tile)) * 512) + off;
      async_copy16((uint4*)sB + i, g);
    }
  };

  v8f acc[NT];
  v8f zero = {0.f, 0.f, 0.f, 0.f, 0.f, 0.f, 0.f, 0.f};
#pragma unroll
  for (int j = 0; j < NT; ++j) acc[j] = zero;

  stage(0, 0);

  for (int kt = 0; kt < Ktiles; ++kt) {
    if (kt + 1 < Ktiles) {
      stage(kt + 1, (kt + 1) & 1);     // prefill next buffer, stays in flight
      asm volatile("s_wait_asynccnt %0" :: "i"(MIN_INFLIGHT) : "memory");
    } else {
      asm volatile("s_wait_asynccnt 0x0" ::: "memory");
    }
    // warm L2->WGP path two K-steps ahead while wmma runs
    if (kt + 2 < Ktiles) {
      const char* nb = (const char*)(Bp + ((size_t)(kt + 2) * NtPadB + ntBase) * 512);
      __builtin_prefetch(nb + (size_t)tid * 64, 0, 0);
    }
    __syncthreads();                    // older batch landed for ALL waves

    const unsigned short* sA = sm[kt & 1];
    const unsigned short* sB = sA + WM * 512;
    v16bf a = *(const v16bf*)(sA + (size_t)wm * 512 + (size_t)lane * 16);
#pragma unroll
    for (int j = 0; j < NT; ++j) {
      v16bf b = *(const v16bf*)(sB + ((size_t)(wn * NT + j)) * 512 + (size_t)lane * 16);
      acc[j] = __builtin_amdgcn_wmma_f32_16x16x32_bf16(
          /*neg_a=*/false, a, /*neg_b=*/false, b,
          /*c_mod=*/(short)0, acc[j], /*reuse_a=*/false, /*reuse_b=*/false);
    }
    __syncthreads();                    // all reads done before buffer reuse
  }

  // epilogue: C layout -> lane = n%16 + 16*hi ; vgpr r -> m = r + 8*hi
  const int hi = lane >> 4, nn = lane & 15;
#pragma unroll
  for (int j = 0; j < NT; ++j) {
    int n = (ntBase + wn * NT + j) * 16 + nn;
    if (n >= Nreal) continue;
    float bv = bias ? bias[n] : 0.f;
#pragma unroll
    for (int r = 0; r < 8; ++r) {
      int m = (mt0 + wm) * 16 + r + 8 * hi;
      float v = acc[j][r] + bv;
      if (addend) v += addend[(size_t)m * ldC + n];
      if (doRelu) v = fmaxf(v, 0.f);
      C[(size_t)m * ldC + n] = v;
    }
  }
}

// ---------------------------------------------------------------------------
// LSTM gate nonlinearity: g(32x4096) -> h,c ; writes h fp32 + bf16 A-packed
// ---------------------------------------------------------------------------
__global__ void lstm_gates(const float* __restrict__ g, float* __restrict__ c,
                           float* __restrict__ hOut, unsigned short* __restrict__ hPack,
                           int packKtiles, int packKtOff) {
  int idx = blockIdx.x * blockDim.x + threadIdx.x;
  if (idx >= 32 * 1024) return;
  int b = idx >> 10, j = idx & 1023;
  const float* gb = g + (size_t)b * 4096;
  float gi = gb[j], gf = gb[1024 + j], gg = gb[2048 + j], go = gb[3072 + j];
  float si = 1.f / (1.f + __expf(-gi));
  float sf = 1.f / (1.f + __expf(-gf));
  float so = 1.f / (1.f + __expf(-go));
  float cn = sf * c[idx] + si * tanhf(gg);
  float h  = so * tanhf(cn);
  c[idx]    = cn;
  hOut[idx] = h;
  hPack[apack_off(b, packKtOff * 32 + j, packKtiles)] = f2bf(h);
}

// ---------------------------------------------------------------------------
// Location-sensitive attention step. One block (128 threads) per batch row.
// ---------------------------------------------------------------------------
__global__ __launch_bounds__(128)
void attention_step(const float* __restrict__ h1,        // 32x1024
                    const float* __restrict__ Wq,        // 1024x128
                    const float* __restrict__ attB,      // 128
                    const float* __restrict__ encp,      // 32x128x128
                    const float* __restrict__ conv,      // 31x128
                    const float* __restrict__ attV,      // 128
                    const unsigned char* __restrict__ mask, // 32x128
                    const float* __restrict__ enc,       // (128,32,512)
                    const float* __restrict__ initCum,   // 32
                    float* __restrict__ cum,             // 32x128
                    float* __restrict__ ctxOut,          // 32x512
                    float* __restrict__ alignsOut,       // 32x128
                    unsigned short* __restrict__ x1pack) // ctx region Kt0..15
{
  const int b = blockIdx.x;
  const int tid = threadIdx.x;
  __shared__ float q[128], cumv[128], red[128], aS[128];

  // q = h1 @ Wq + att_b
  float acc = attB[tid];
  const float* hb = h1 + (size_t)b * 1024;
  for (int k = 0; k < 1024; ++k) acc += hb[k] * Wq[(size_t)k * 128 + tid];
  q[tid] = acc;
  cumv[tid] = cum[b * 128 + tid] + initCum[b];
  __syncthreads();

  // energies (thread = token position t)
  float e;
  if (mask[b * 128 + tid]) {
    e = 0.f;
    for (int a = 0; a < 128; ++a) {
      float loc = 0.f;
#pragma unroll
      for (int k = 0; k < 31; ++k) {
        int tt = tid + k - 15;
        if (tt >= 0 && tt < 128) loc += cumv[tt] * conv[k * 128 + a];
      }
      float feat = tanhf(q[a] + encp[((size_t)b * 128 + tid) * 128 + a] + loc);
      e += feat * attV[a];
    }
  } else {
    e = -1e9f;
  }

  // softmax over 128 tokens
  red[tid] = e; __syncthreads();
  for (int s = 64; s > 0; s >>= 1) { if (tid < s) red[tid] = fmaxf(red[tid], red[tid + s]); __syncthreads(); }
  float mx = red[0]; __syncthreads();
  float ex = __expf(e - mx);
  red[tid] = ex; __syncthreads();
  for (int s = 64; s > 0; s >>= 1) { if (tid < s) red[tid] += red[tid + s]; __syncthreads(); }
  float at = ex / red[0];
  aS[tid] = at;
  alignsOut[(size_t)b * 128 + tid] = at;
  cum[b * 128 + tid] = (cumv[tid] - initCum[b]) + at;
  __syncthreads();

  // ctx = a @ enc ; also bf16-pack into next step's A operand (Kt 0..15)
  for (int ei = tid; ei < 512; ei += 128) {
    float s = 0.f;
    for (int tt = 0; tt < 128; ++tt)
      s += aS[tt] * enc[((size_t)tt * 32 + b) * 512 + ei];
    ctxOut[(size_t)b * 512 + ei] = s;
    x1pack[apack_off(b, ei, 48)] = f2bf(s);
  }
}

// ---------------------------------------------------------------------------
// enc_proj[b,t,a] = enc[b,t,:] @ att_We  (one-time, fp32)
// ---------------------------------------------------------------------------
__global__ void enc_proj_k(const float* __restrict__ enc, const float* __restrict__ We,
                           float* __restrict__ out) {
  int idx = blockIdx.x * blockDim.x + threadIdx.x;
  if (idx >= 32 * 128 * 128) return;
  int a = idx & 127, bt = idx >> 7;
  int tt = bt & 127, b = bt >> 7;
  const float* e = enc + ((size_t)tt * 32 + b) * 512;
  float s = 0.f;
  for (int k = 0; k < 512; ++k) s += e[k] * We[(size_t)k * 128 + a];
  out[((size_t)b * 128 + tt) * 128 + a] = s;
}

// ---------------------------------------------------------------------------
// init MLP (tiny, fp32)
// ---------------------------------------------------------------------------
__global__ void init_mlp1(const float* __restrict__ spk, const float* __restrict__ enc0,
                          const float* __restrict__ W1, const float* __restrict__ b1,
                          float* __restrict__ hbuf) {
  int idx = blockIdx.x * blockDim.x + threadIdx.x;
  if (idx >= 32 * 640) return;
  int b = idx / 640, j = idx % 640;
  float s = b1[j];
  for (int k = 0; k < 128; ++k) s += spk[b * 128 + k] * W1[(size_t)k * 640 + j];
  for (int k = 0; k < 512; ++k) s += enc0[(size_t)b * 512 + k] * W1[(size_t)(128 + k) * 640 + j];
  hbuf[idx] = fmaxf(s, 0.f);
}

__global__ void init_mlp2(const float* __restrict__ hbuf, const float* __restrict__ W2,
                          const float* __restrict__ b2, float* __restrict__ frame0,
                          float* __restrict__ initCum, unsigned short* __restrict__ x1pack) {
  int idx = blockIdx.x * blockDim.x + threadIdx.x;
  if (idx >= 32 * 593) return;
  int b = idx / 593, j = idx % 593;
  const float* hb = hbuf + (size_t)b * 640;
  float s = b2[j];
  for (int k = 0; k < 640; ++k) s += hb[k] * W2[(size_t)k * 593 + j];
  if (j < 80)       frame0[b * 80 + j] = s;
  else if (j == 80) initCum[b] = fabsf(s);
  else              x1pack[apack_off(b, j - 81, 48)] = f2bf(s);
}

// stop_token = h2s @ stop_W + stop_b
__global__ void stop_k(const float* __restrict__ h2s, const float* __restrict__ W,
                       const float* __restrict__ bb, float* __restrict__ out) {
  int idx = blockIdx.x * blockDim.x + threadIdx.x;
  if (idx >= 19200) return;
  const float* h = h2s + (size_t)idx * 1024;
  float s = bb[0];
  for (int k = 0; k < 1024; ++k) s += h[k] * W[k];
  out[idx] = s;
}

// ---------------------------------------------------------------------------
// Host driver
// ---------------------------------------------------------------------------
#define CEILDIV(a, b) (((a) + (b) - 1) / (b))

extern "C" void kernel_launch(void* const* d_in, const int* in_sizes, int n_in,
                              void* d_out, int out_size, void* d_ws, size_t ws_size,
                              hipStream_t stream) {
  (void)in_sizes; (void)n_in; (void)out_size; (void)ws_size;

  const float* encoded  = (const float*)d_in[0];   // (128,32,512)
  const unsigned char* mask = (const unsigned char*)d_in[1]; // (32,128)
  const float* speaker  = (const float*)d_in[2];   // (32,128)
  const float* tgt      = (const float*)d_in[4];   // (600,32,80)
  const float* init_W1  = (const float*)d_in[5];
  const float* init_b1  = (const float*)d_in[6];
  const float* init_W2  = (const float*)d_in[7];
  const float* init_b2  = (const float*)d_in[8];
  const float* pre_W1   = (const float*)d_in[9];
  const float* pre_b1   = (const float*)d_in[10];
  const float* pre_W2   = (const float*)d_in[11];
  const float* pre_b2   = (const float*)d_in[12];
  const float* l1_Wih   = (const float*)d_in[13];  // (896,4096)
  const float* l1_Whh   = (const float*)d_in[14];  // (1024,4096)
  const float* l1_b     = (const float*)d_in[15];
  const float* l2_Wih   = (const float*)d_in[16];  // (1664,4096)
  const float* l2_Whh   = (const float*)d_in[17];
  const float* l2_b     = (const float*)d_in[18];
  const float* att_Wq   = (const float*)d_in[19];
  const float* att_We   = (const float*)d_in[20];
  const float* att_b    = (const float*)d_in[21];
  const float* att_conv = (const float*)d_in[22];
  const float* att_v    = (const float*)d_in[23];
  const float* out_W    = (const float*)d_in[24];
  const float* out_b    = (const float*)d_in[25];
  const float* stop_W   = (const float*)d_in[26];
  const float* stop_b   = (const float*)d_in[27];

  float* out_frames = (float*)d_out;                 // 600*32*80
  float* out_stop   = out_frames + 600 * 32 * 80;    // 600*32
  float* out_aligns = out_stop + 600 * 32;           // 600*32*128

  // ---- workspace bump allocator ----
  size_t off = 0;
  auto alloc = [&](size_t bytes) -> char* {
    char* p = (char*)d_ws + off;
    off += (bytes + 255) & ~(size_t)255;
    return p;
  };
  unsigned short* preW1p = (unsigned short*)alloc((size_t)3 * 16 * 512 * 2);
  unsigned short* preW2p = (unsigned short*)alloc((size_t)8 * 16 * 512 * 2);
  unsigned short* W1fp   = (unsigned short*)alloc((size_t)12 * 256 * 512 * 2);
  unsigned short* W1sp   = (unsigned short*)alloc((size_t)48 * 256 * 512 * 2);
  unsigned short* W2ihp  = (unsigned short*)alloc((size_t)52 * 256 * 512 * 2);
  unsigned short* W2hhp  = (unsigned short*)alloc((size_t)32 * 256 * 512 * 2);
  unsigned short* outWp  = (unsigned short*)alloc((size_t)52 * 8 * 512 * 2);
  unsigned short* FINp = (unsigned short*)alloc((size_t)1200 * 3 * 512 * 2);
  unsigned short* P1p  = (unsigned short*)alloc((size_t)1200 * 8 * 512 * 2);
  unsigned short* PNp  = (unsigned short*)alloc((size_t)1200 * 12 * 512 * 2);
  unsigned short* X2p  = (unsigned short*)alloc((size_t)1200 * 52 * 512 * 2);
  unsigned short* x1p  = (unsigned short*)alloc((size_t)2 * 48 * 512 * 2);
  unsigned short* x2p  = (unsigned short*)alloc((size_t)2 * 32 * 512 * 2);
  float* p1    = (float*)alloc((size_t)19200 * 256 * 4);
  float* pn    = (float*)alloc((size_t)19200 * 256 * 4);
  float* pc1   = (float*)alloc((size_t)19200 * 4096 * 4);
  float* pc2   = (float*)alloc((size_t)19200 * 4096 * 4);
  float* hs    = (float*)alloc((size_t)19200 * 1024 * 4);
  float* h2s   = (float*)alloc((size_t)19200 * 1024 * 4);
  float* ctxs  = (float*)alloc((size_t)19200 * 512 * 4);
  float* gbuf  = (float*)alloc((size_t)32 * 4096 * 4);
  float* c1    = (float*)alloc((size_t)32 * 1024 * 4);
  float* c2    = (float*)alloc((size_t)32 * 1024 * 4);
  float* cum   = (float*)alloc((size_t)32 * 128 * 4);
  float* initC = (float*)alloc((size_t)32 * 4);
  float* hbuf  = (float*)alloc((size_t)32 * 640 * 4);
  float* fr0   = (float*)alloc((size_t)32 * 80 * 4);
  float* encp  = (float*)alloc((size_t)32 * 128 * 128 * 4);

  // ---- zero state (capturable async memsets) ----
  hipMemsetAsync(x1p, 0, (size_t)2 * 48 * 512 * 2, stream); // h region 0 for step 0
  hipMemsetAsync(x2p, 0, (size_t)2 * 32 * 512 * 2, stream);
  hipMemsetAsync(c1, 0, (size_t)32 * 1024 * 4, stream);
  hipMemsetAsync(c2, 0, (size_t)32 * 1024 * 4, stream);
  hipMemsetAsync(cum, 0, (size_t)32 * 128 * 4, stream);

  // ---- pack weights to WMMA bf16 tiles ----
  auto packW = [&](const float* src, int rowOff, int Kc, int Nc, int ldN,
                   unsigned short* dst, int NtPad, int ktOff, int Kpad) {
    size_t tot = (size_t)Kpad * NtPad * 16;
    pack_weight<<<(unsigned)CEILDIV(tot, 256), 256, 0, stream>>>(
        src, rowOff, Kc, Nc, ldN, dst, NtPad, ktOff, Kpad);
  };
  packW(pre_W1, 0, 80, 256, 256, preW1p, 16, 0, 96);
  packW(pre_W2, 0, 256, 256, 256, preW2p, 16, 0, 256);
  packW(l1_Wih, 0, 256, 4096, 4096, W1fp, 256, 0, 256);   // prenet rows
  packW(l1_Wih, 768, 128, 4096, 4096, W1fp, 256, 8, 128); // speaker rows
  packW(l1_Wih, 256, 512, 4096, 4096, W1sp, 256, 0, 512); // ctx rows (Kt0..15)
  packW(l1_Whh, 0, 1024, 4096, 4096, W1sp, 256, 16, 1024);// h rows (Kt16..47)
  packW(l2_Wih, 0, 1664, 4096, 4096, W2ihp, 256, 0, 1664);
  packW(l2_Whh, 0, 1024, 4096, 4096, W2hhp, 256, 0, 1024);
  packW(out_W, 0, 1664, 80, 80, outWp, 8, 0, 1664);

  // ---- init MLP + enc_proj ----
  init_mlp1<<<CEILDIV(32 * 640, 256), 256, 0, stream>>>(speaker, encoded, init_W1, init_b1, hbuf);
  init_mlp2<<<CEILDIV(32 * 593, 256), 256, 0, stream>>>(hbuf, init_W2, init_b2, fr0, initC, x1p);
  enc_proj_k<<<CEILDIV(32 * 128 * 128, 256), 256, 0, stream>>>(encoded, att_We, encp);

  auto packA = [&](const float* src, int srcLd, int Kc, int Kpad, unsigned short* dst,
                   int Ktiles, int ktOff, int Mc, int mOff, int rowMod) {
    size_t tot = (size_t)Mc * Kpad;
    pack_act<<<(unsigned)CEILDIV(tot, 256), 256, 0, stream>>>(
        src, srcLd, Kc, Kpad, dst, Ktiles, ktOff, Mc, mOff, rowMod);
  };

  // ---- prenet over all 600 steps (teacher forcing) ----
  packA(fr0, 80, 80, 96, FINp, 3, 0, 32, 0, 0);
  packA(tgt, 80, 80, 96, FINp, 3, 0, 19168, 32, 0);
  gemm_bf16<4, 2, 4><<<dim3(2, 300), 256, 0, stream>>>(FINp, preW1p, 3, 16, p1, 256, 256, pre_b1, nullptr, 1);
  packA(p1, 256, 256, 256, P1p, 8, 0, 19200, 0, 0);
  gemm_bf16<4, 2, 4><<<dim3(2, 300), 256, 0, stream>>>(P1p, preW2p, 8, 16, pn, 256, 256, pre_b2, nullptr, 1);

  // ---- precomp1 = [pn, spk] @ W1f + l1_b ----
  packA(pn, 256, 256, 256, PNp, 12, 0, 19200, 0, 0);
  packA(speaker, 128, 128, 128, PNp, 12, 8, 19200, 0, 32);
  gemm_bf16<4, 2, 4><<<dim3(32, 300), 256, 0, stream>>>(PNp, W1fp, 12, 256, pc1, 4096, 4096, l1_b, nullptr, 0);

  // ---- scan 1: LSTM1 + attention (sequential, 600 steps) ----
  for (int t = 0; t < 600; ++t) {
    gemm_bf16<2, 4, 4><<<dim3(16, 1), 256, 0, stream>>>(
        x1p, W1sp, 48, 256, gbuf, 4096, 4096, nullptr, pc1 + (size_t)t * 32 * 4096, 0);
    lstm_gates<<<CEILDIV(32 * 1024, 256), 256, 0, stream>>>(
        gbuf, c1, hs + (size_t)t * 32 * 1024, x1p, 48, 16);
    attention_step<<<32, 128, 0, stream>>>(
        hs + (size_t)t * 32 * 1024, att_Wq, att_b, encp, att_conv, att_v, mask,
        encoded, initC, cum, ctxs + (size_t)t * 32 * 512,
        out_aligns + (size_t)t * 32 * 128, x1p);
  }

  // ---- precomp2 = [h1s, ctxs, spk] @ l2_Wih + l2_b (one big parallel GEMM) ----
  packA(hs, 1024, 1024, 1024, X2p, 52, 0, 19200, 0, 0);
  packA(ctxs, 512, 512, 512, X2p, 52, 32, 19200, 0, 0);
  packA(speaker, 128, 128, 128, X2p, 52, 48, 19200, 0, 32);
  gemm_bf16<4, 2, 4><<<dim3(32, 300), 256, 0, stream>>>(X2p, W2ihp, 52, 256, pc2, 4096, 4096, l2_b, nullptr, 0);

  // ---- scan 2: LSTM2 recurrence ----
  for (int t = 0; t < 600; ++t) {
    gemm_bf16<2, 4, 4><<<dim3(16, 1), 256, 0, stream>>>(
        x2p, W2hhp, 32, 256, gbuf, 4096, 4096, nullptr, pc2 + (size_t)t * 32 * 4096, 0);
    lstm_gates<<<CEILDIV(32 * 1024, 256), 256, 0, stream>>>(
        gbuf, c2, h2s + (size_t)t * 32 * 1024, x2p, 32, 0);
  }

  // ---- outputs: frames = [h2s, ctxs, spk] @ out_W + out_b ; stop ----
  packA(h2s, 1024, 1024, 1024, X2p, 52, 0, 19200, 0, 0); // overwrite h1 part with h2
  gemm_bf16<4, 2, 4><<<dim3(1, 300), 256, 0, stream>>>(X2p, outWp, 52, 8, out_frames, 80, 80, out_b, nullptr, 0);
  stop_k<<<CEILDIV(19200, 256), 256, 0, stream>>>(h2s, stop_W, stop_b, out_stop);
}
